// MambaBlock_36309653520717
// MI455X (gfx1250) — matrix operand order repaired and checked by reading, not compiled, not run on hardware
//
#include <hip/hip_runtime.h>
#include <hip/hip_bf16.h>
#include <cstddef>
#include <cstdint>

// ---------------------------------------------------------------------------
// Types for CDNA5 WMMA (wave32): v_wmma_f32_16x16x32_f16
// ---------------------------------------------------------------------------
typedef _Float16 v16h __attribute__((ext_vector_type(16)));
typedef _Float16 v8h  __attribute__((ext_vector_type(8)));
typedef float    v8f  __attribute__((ext_vector_type(8)));

// ---------------------------------------------------------------------------
// Problem dimensions (fixed by the reference)
// ---------------------------------------------------------------------------
static constexpr int Bc  = 16;          // batch
static constexpr int Lc  = 256;         // seq len
static constexpr int Dmc = 512;         // d_model
static constexpr int DIc = 1024;        // d_inner
static constexpr int Mc  = Bc * Lc;     // 4096 token rows
static constexpr int N2c = 2 * DIc;     // 2048
static constexpr int Rc  = 32;          // dt_rank
static constexpr int Nsc = 16;          // d_state
static constexpr int P1c = Rc + 2 * Nsc;   // 64   (dr | B | C)
static constexpr int P3c = Rc + 2 * DIc;   // 2080 (dr_s | Bs | Cs)

// ===========================================================================
// f32 -> f16 conversion (optionally strided source rows)
// ===========================================================================
__global__ void cvt_f16_k(const float* __restrict__ src, _Float16* __restrict__ dst,
                          int rows, int cols, int srcLd, int srcOff) {
  int idx = blockIdx.x * blockDim.x + threadIdx.x;
  if (idx >= rows * cols) return;
  int r = idx / cols, c = idx - r * cols;
  dst[idx] = (_Float16)src[(size_t)r * srcLd + srcOff + c];
}

// f32 (K x N) -> f16 transposed (N x K)
__global__ void cvt_f16_T_k(const float* __restrict__ src, _Float16* __restrict__ dst,
                            int K, int N) {
  int idx = blockIdx.x * blockDim.x + threadIdx.x;
  if (idx >= K * N) return;
  int n = idx / K, k = idx - n * K;
  dst[(size_t)n * K + k] = (_Float16)src[(size_t)k * N + n];
}

// ===========================================================================
// WMMA GEMM:  C[M x N] (f32, ldc=N) = A[M x K] (f16) * BT[N x K] (f16, B^T).
// Wave tile = 64x32: 4 A-fragments x 2 B-fragments -> 8 v_wmma per K=32 step.
// Requires M % 64 == 0, N % 32 == 0, K % 32 == 0 (true for all calls here).
// mode 1: epilogue out = softplus(acc + bias[col]).
// ===========================================================================
__global__ void gemm_wmma_f16_k(const _Float16* __restrict__ A,
                                const _Float16* __restrict__ BT,
                                float* __restrict__ C, int M, int N, int K,
                                const float* __restrict__ bias, int mode) {
  const int lane = threadIdx.x & 31;
  const int wave = threadIdx.x >> 5;
  const int tilesN = N >> 5;
  const int tilesM = M >> 6;
  int tile = blockIdx.x * 8 + wave;
  if (tile >= tilesM * tilesN) return;
  const int tm = (tile / tilesN) << 6;
  const int tn = (tile % tilesN) << 5;

  const int r16  = lane & 15;
  const int hi   = lane >> 4;       // 0: lanes 0-15, 1: lanes 16-31
  const int aoff = hi * 16;         // A K-sub offset (contiguous, matches B)
  const int boff = hi * 16;         // B K-sub offset (contiguous, matches A)

  v8f acc[8];
#pragma unroll
  for (int i = 0; i < 8; ++i) acc[i] = (v8f){};

  const _Float16* a0p = A + (size_t)(tm +  0 + r16) * K + aoff;
  const _Float16* a1p = A + (size_t)(tm + 16 + r16) * K + aoff;
  const _Float16* a2p = A + (size_t)(tm + 32 + r16) * K + aoff;
  const _Float16* a3p = A + (size_t)(tm + 48 + r16) * K + aoff;
  const _Float16* b0p = BT + (size_t)(tn      + r16) * K + boff;
  const _Float16* b1p = BT + (size_t)(tn + 16 + r16) * K + boff;

  for (int kk = 0; kk < K; kk += 32) {
    v16h a0, a1, a2, a3;
    {
      v8h l0 = *(const v8h*)(a0p + kk), h0 = *(const v8h*)(a0p + kk + 8);
      v8h l1 = *(const v8h*)(a1p + kk), h1 = *(const v8h*)(a1p + kk + 8);
      v8h l2 = *(const v8h*)(a2p + kk), h2 = *(const v8h*)(a2p + kk + 8);
      v8h l3 = *(const v8h*)(a3p + kk), h3 = *(const v8h*)(a3p + kk + 8);
#pragma unroll
      for (int j = 0; j < 8; ++j) {
        a0[j] = l0[j]; a0[j + 8] = h0[j];
        a1[j] = l1[j]; a1[j + 8] = h1[j];
        a2[j] = l2[j]; a2[j + 8] = h2[j];
        a3[j] = l3[j]; a3[j + 8] = h3[j];
      }
    }
    v16h b0 = *(const v16h*)(b0p + kk);
    v16h b1 = *(const v16h*)(b1p + kk);
    acc[0] = __builtin_amdgcn_wmma_f32_16x16x32_f16(false, a0, false, b0, (short)0, acc[0], false, false);
    acc[1] = __builtin_amdgcn_wmma_f32_16x16x32_f16(false, a0, false, b1, (short)0, acc[1], false, false);
    acc[2] = __builtin_amdgcn_wmma_f32_16x16x32_f16(false, a1, false, b0, (short)0, acc[2], false, false);
    acc[3] = __builtin_amdgcn_wmma_f32_16x16x32_f16(false, a1, false, b1, (short)0, acc[3], false, false);
    acc[4] = __builtin_amdgcn_wmma_f32_16x16x32_f16(false, a2, false, b0, (short)0, acc[4], false, false);
    acc[5] = __builtin_amdgcn_wmma_f32_16x16x32_f16(false, a2, false, b1, (short)0, acc[5], false, false);
    acc[6] = __builtin_amdgcn_wmma_f32_16x16x32_f16(false, a3, false, b0, (short)0, acc[6], false, false);
    acc[7] = __builtin_amdgcn_wmma_f32_16x16x32_f16(false, a3, false, b1, (short)0, acc[7], false, false);
  }

  // C/D layout: VGPR v -> row v (+8 for lanes 16-31), col = lane&15
  const int cc = tn + r16;
  const float b0v = (mode == 1) ? bias[cc]      : 0.f;
  const float b1v = (mode == 1) ? bias[cc + 16] : 0.f;
#pragma unroll
  for (int mi = 0; mi < 4; ++mi) {
    const int cr = tm + mi * 16 + hi * 8;
#pragma unroll
    for (int v = 0; v < 8; ++v) {
      float x0 = acc[mi * 2 + 0][v];
      float x1 = acc[mi * 2 + 1][v];
      if (mode == 1) {
        x0 += b0v; x0 = (x0 > 20.f) ? x0 : log1pf(__expf(x0));
        x1 += b1v; x1 = (x1 > 20.f) ? x1 : log1pf(__expf(x1));
      }
      C[(size_t)(cr + v) * N + cc]      = x0;
      C[(size_t)(cr + v) * N + cc + 16] = x1;
    }
  }
}

// ===========================================================================
// Causal depthwise conv (K=4) + SiLU; writes both f32 and f16 copies of u.
// ===========================================================================
__global__ void conv_silu_k(const float* __restrict__ xz,
                            const float* __restrict__ conv_w,
                            const float* __restrict__ conv_b,
                            float* __restrict__ u, _Float16* __restrict__ u16) {
  int idx = blockIdx.x * blockDim.x + threadIdx.x;
  if (idx >= Mc * DIc) return;
  int d = idx & (DIc - 1);
  int m = idx >> 10;
  int l = m & (Lc - 1);
  float acc = conv_b[d];
#pragma unroll
  for (int k = 0; k < 4; ++k) {
    int ls = l + k - 3;
    if (ls >= 0)
      acc += xz[(size_t)(m + k - 3) * N2c + d] * conv_w[d * 4 + k];
  }
  float s = acc / (1.0f + __expf(-acc));   // silu
  u[idx]   = s;
  u16[idx] = (_Float16)s;
}

// tiny dense matmul: out[Bn x Nn] = A[Bn x Kn] * W[Kn x Nn]
__global__ void small_mm_k(const float* __restrict__ A, const float* __restrict__ W,
                           float* __restrict__ out, int Bn, int Kn, int Nn) {
  int idx = blockIdx.x * blockDim.x + threadIdx.x;
  if (idx >= Bn * Nn) return;
  int b = idx / Nn, n = idx - b * Nn;
  float acc = 0.f;
  for (int k = 0; k < Kn; ++k) acc += A[(size_t)b * Kn + k] * W[(size_t)k * Nn + n];
  out[idx] = acc;
}

// y = res   (res = xz[:, :, DI:2*DI])
__global__ void init_y_k(const float* __restrict__ xz, float* __restrict__ y) {
  int idx = blockIdx.x * blockDim.x + threadIdx.x;
  if (idx >= Mc * DIc) return;
  int d = idx & (DIc - 1);
  int m = idx >> 10;
  y[idx] = xz[(size_t)m * N2c + DIc + d];
}

// ===========================================================================
// Temporal selective scan (fwd or bwd). One thread per (b,d); 16-state in
// registers; per-timestep B/C staged in LDS (shared by all 1024 channels).
// ===========================================================================
__global__ void temporal_scan_k(const float* __restrict__ delta,
                                const float* __restrict__ pBC,   // [M x 64]: dr|B|C
                                const float* __restrict__ A_log,
                                const float* __restrict__ Dvec,
                                const float* __restrict__ u,
                                const float* __restrict__ kp1,   // [B x DI]
                                const float* __restrict__ key2,  // [16]
                                float* __restrict__ y,
                                int reverse) {
  const int b = blockIdx.y;
  const int d = blockIdx.x * blockDim.x + threadIdx.x;
  __shared__ float sB[Nsc];
  __shared__ float sC[Nsc];

  float An[Nsc], s[Nsc];
  const float s0v = kp1[(size_t)b * DIc + d];
#pragma unroll
  for (int n = 0; n < Nsc; ++n) {
    An[n] = -__expf(A_log[n]);
    s[n]  = s0v * key2[n];
  }
  const float Dv = Dvec[d];

  for (int i = 0; i < Lc; ++i) {
    const int t = reverse ? (Lc - 1 - i) : i;
    const size_t mt = (size_t)b * Lc + t;
    __syncthreads();
    if (threadIdx.x < 32) {
      float v = pBC[mt * P1c + Rc + threadIdx.x];
      if (threadIdx.x < Nsc) sB[threadIdx.x] = v;
      else                   sC[threadIdx.x - Nsc] = v;
    }
    __syncthreads();
    const float dl = delta[mt * DIc + d];
    const float uv = u[mt * DIc + d];
    const float du = dl * uv;
    float yacc = 0.f;
#pragma unroll
    for (int n = 0; n < Nsc; ++n) {
      float dA = __expf(dl * An[n]);
      s[n] = dA * s[n] + du * sB[n];
      yacc += s[n] * sC[n];
    }
    y[mt * DIc + d] += yacc + uv * Dv;
  }
}

// per-token reductions for the spatial scan: S = sum_d delta_s, T = sum_d delta_s*u
__global__ void rowsum_k(const float* __restrict__ ds, const float* __restrict__ u,
                         float* __restrict__ S, float* __restrict__ T) {
  const int m = blockIdx.x;
  __shared__ float rs[256];
  __shared__ float rt[256];
  float a = 0.f, t = 0.f;
  for (int d = threadIdx.x; d < DIc; d += 256) {
    float v = ds[(size_t)m * DIc + d];
    a += v;
    t += v * u[(size_t)m * DIc + d];
  }
  rs[threadIdx.x] = a; rt[threadIdx.x] = t;
  __syncthreads();
  for (int off = 128; off > 0; off >>= 1) {
    if (threadIdx.x < off) { rs[threadIdx.x] += rs[threadIdx.x + off];
                             rt[threadIdx.x] += rt[threadIdx.x + off]; }
    __syncthreads();
  }
  if (threadIdx.x == 0) { S[m] = rs[0]; T[m] = rt[0]; }
}

// ===========================================================================
// Spatial scan: channel recurrence, state per token. Bs/Cs chunks are staged
// through LDS with 32B-run cooperative loads (pad 9 -> conflict-free reads).
// Produces the final y in f16 directly (y16 = yin + s*Cs + u*D_s).
// ===========================================================================
__global__ void spatial_scan_k(const float* __restrict__ S, const float* __restrict__ T,
                               const float* __restrict__ p3,      // [M x 2080]
                               const float* __restrict__ As_log,  // [DI]
                               const float* __restrict__ Ds,
                               const float* __restrict__ keys0,   // [M]
                               const float* __restrict__ u,
                               const float* __restrict__ yin,     // res + yf + yb
                               _Float16* __restrict__ y16) {
  const int tid = threadIdx.x;
  const int m0  = blockIdx.x * 256;
  const int m   = m0 + tid;
  __shared__ float sAs[DIc];
  __shared__ float sB[256 * 9];
  __shared__ float sC[256 * 9];
  for (int d = tid; d < DIc; d += 256) sAs[d] = -__expf(As_log[d]);

  float s = keys0[m];
  const float Sv = S[m], Tv = T[m];

  for (int d0 = 0; d0 < DIc; d0 += 8) {
    __syncthreads();
    for (int e = tid; e < 256 * 8; e += 256) {
      int r = e >> 3, c = e & 7;
      size_t base = (size_t)(m0 + r) * P3c + d0 + c;
      sB[r * 9 + c] = p3[base + Rc];          // Bs
      sC[r * 9 + c] = p3[base + Rc + DIc];    // Cs
    }
    __syncthreads();
#pragma unroll
    for (int j = 0; j < 8; ++j) {
      const int d = d0 + j;
      float a = __expf(Sv * sAs[d]);
      s = a * s + Tv * sB[tid * 9 + j];
      const size_t idx = (size_t)m * DIc + d;
      y16[idx] = (_Float16)(yin[idx] + s * sC[tid * 9 + j] + u[idx] * Ds[d]);
    }
  }
}

// ===========================================================================
// Host: pipeline on `stream`, scratch carved from d_ws
// ===========================================================================
static inline size_t alignup(size_t x) { return (x + 255) & ~(size_t)255; }
static inline int cdiv(int a, int b) { return (a + b - 1) / b; }

extern "C" void kernel_launch(void* const* d_in, const int* in_sizes, int n_in,
                              void* d_out, int out_size, void* d_ws, size_t ws_size,
                              hipStream_t stream) {
  const float* x         = (const float*)d_in[0];
  const float* x_key     = (const float*)d_in[1];
  const float* in_proj_w = (const float*)d_in[2];
  const float* conv_w    = (const float*)d_in[3];
  const float* conv_b    = (const float*)d_in[4];
  const float* x_proj_w  = (const float*)d_in[5];
  const float* x_proj_bw = (const float*)d_in[6];
  const float* x_proj_sw = (const float*)d_in[7];
  const float* dt_w      = (const float*)d_in[8];
  const float* dt_bias   = (const float*)d_in[9];
  const float* dt_b_w    = (const float*)d_in[10];
  const float* dt_b_bias = (const float*)d_in[11];
  const float* dt_s_w    = (const float*)d_in[12];
  const float* dt_s_bias = (const float*)d_in[13];
  const float* key1_w    = (const float*)d_in[14];
  const float* key2_w    = (const float*)d_in[15];
  const float* keys_w    = (const float*)d_in[16];
  const float* A_log     = (const float*)d_in[17];
  const float* A_b_log   = (const float*)d_in[18];
  const float* A_s_log   = (const float*)d_in[19];
  const float* Dv        = (const float*)d_in[20];
  const float* D_b       = (const float*)d_in[21];
  const float* D_s       = (const float*)d_in[22];
  const float* out_proj_w= (const float*)d_in[23];
  float* out = (float*)d_out;

  // ---- workspace carving ----
  char* p = (char*)d_ws;
  auto alloc = [&](size_t bytes) { void* r = (void*)p; p += alignup(bytes); return r; };

  float*    xz      = (float*)alloc((size_t)Mc * N2c * 4);
  float*    u       = (float*)alloc((size_t)Mc * DIc * 4);
  float*    p1      = (float*)alloc((size_t)Mc * P1c * 4);
  float*    p2      = (float*)alloc((size_t)Mc * P1c * 4);
  float*    p3      = (float*)alloc((size_t)Mc * P3c * 4);
  float*    delta   = (float*)alloc((size_t)Mc * DIc * 4);
  float*    delta_b = (float*)alloc((size_t)Mc * DIc * 4);
  float*    delta_s = (float*)alloc((size_t)Mc * DIc * 4);
  float*    y       = (float*)alloc((size_t)Mc * DIc * 4);
  float*    kp1     = (float*)alloc((size_t)Bc * DIc * 4);
  float*    keys0   = (float*)alloc((size_t)Bc * Lc * 4);
  float*    Ssum    = (float*)alloc((size_t)Mc * 4);
  float*    Tsum    = (float*)alloc((size_t)Mc * 4);

  _Float16* x16     = (_Float16*)alloc((size_t)Mc * Dmc * 2);
  _Float16* win16   = (_Float16*)alloc((size_t)N2c * Dmc * 2);     // in_proj^T
  _Float16* u16     = (_Float16*)alloc((size_t)Mc * DIc * 2);
  _Float16* wxp16   = (_Float16*)alloc((size_t)P1c * DIc * 2);
  _Float16* wxpb16  = (_Float16*)alloc((size_t)P1c * DIc * 2);
  _Float16* wxps16  = (_Float16*)alloc((size_t)P3c * DIc * 2);
  _Float16* dr16    = (_Float16*)alloc((size_t)Mc * Rc * 2);
  _Float16* drb16   = (_Float16*)alloc((size_t)Mc * Rc * 2);
  _Float16* drs16   = (_Float16*)alloc((size_t)Mc * Rc * 2);
  _Float16* dtw16   = (_Float16*)alloc((size_t)DIc * Rc * 2);
  _Float16* dtbw16  = (_Float16*)alloc((size_t)DIc * Rc * 2);
  _Float16* dtsw16  = (_Float16*)alloc((size_t)DIc * Rc * 2);
  _Float16* y16     = (_Float16*)alloc((size_t)Mc * DIc * 2);
  _Float16* wout16  = (_Float16*)alloc((size_t)Dmc * DIc * 2);
  (void)ws_size; (void)in_sizes; (void)n_in; (void)out_size;

  const int TB = 256;
  auto gemm = [&](const _Float16* A, const _Float16* BT, float* C,
                  int M, int N, int K, const float* bias, int mode) {
    int tiles = (M >> 6) * (N >> 5);
    gemm_wmma_f16_k<<<cdiv(tiles, 8), TB, 0, stream>>>(A, BT, C, M, N, K, bias, mode);
  };

  // ---- stage inputs to f16 (weights transposed to N x K) ----
  cvt_f16_k  <<<cdiv(Mc * Dmc, TB), TB, 0, stream>>>(x, x16, Mc, Dmc, Dmc, 0);
  cvt_f16_T_k<<<cdiv(Dmc * N2c, TB), TB, 0, stream>>>(in_proj_w, win16, Dmc, N2c);
  cvt_f16_T_k<<<cdiv(DIc * P1c, TB), TB, 0, stream>>>(x_proj_w,  wxp16,  DIc, P1c);
  cvt_f16_T_k<<<cdiv(DIc * P1c, TB), TB, 0, stream>>>(x_proj_bw, wxpb16, DIc, P1c);
  cvt_f16_T_k<<<cdiv(DIc * P3c, TB), TB, 0, stream>>>(x_proj_sw, wxps16, DIc, P3c);
  cvt_f16_T_k<<<cdiv(Rc * DIc, TB), TB, 0, stream>>>(dt_w,   dtw16,  Rc, DIc);
  cvt_f16_T_k<<<cdiv(Rc * DIc, TB), TB, 0, stream>>>(dt_b_w, dtbw16, Rc, DIc);
  cvt_f16_T_k<<<cdiv(Rc * DIc, TB), TB, 0, stream>>>(dt_s_w, dtsw16, Rc, DIc);
  cvt_f16_T_k<<<cdiv(DIc * Dmc, TB), TB, 0, stream>>>(out_proj_w, wout16, DIc, Dmc);

  // ---- xz = x @ in_proj_w ----
  gemm(x16, win16, xz, Mc, N2c, Dmc, nullptr, 0);

  // ---- u = silu(depthwise_conv(xz[:, :, :DI])) (f32 + f16) ----
  conv_silu_k<<<cdiv(Mc * DIc, TB), TB, 0, stream>>>(xz, conv_w, conv_b, u, u16);

  // ---- projections ----
  gemm(u16, wxp16,  p1, Mc, P1c, DIc, nullptr, 0);
  gemm(u16, wxpb16, p2, Mc, P1c, DIc, nullptr, 0);
  gemm(u16, wxps16, p3, Mc, P3c, DIc, nullptr, 0);

  // ---- delta = softplus(dr @ dt_w + bias) fused into GEMM epilogue (x3) ----
  cvt_f16_k<<<cdiv(Mc * Rc, TB), TB, 0, stream>>>(p1, dr16,  Mc, Rc, P1c, 0);
  cvt_f16_k<<<cdiv(Mc * Rc, TB), TB, 0, stream>>>(p2, drb16, Mc, Rc, P1c, 0);
  cvt_f16_k<<<cdiv(Mc * Rc, TB), TB, 0, stream>>>(p3, drs16, Mc, Rc, P3c, 0);
  gemm(dr16,  dtw16,  delta,   Mc, DIc, Rc, dt_bias,   1);
  gemm(drb16, dtbw16, delta_b, Mc, DIc, Rc, dt_b_bias, 1);
  gemm(drs16, dtsw16, delta_s, Mc, DIc, Rc, dt_s_bias, 1);

  // ---- initial states ----
  small_mm_k<<<cdiv(Bc * DIc, TB), TB, 0, stream>>>(x_key, key1_w, kp1,   Bc, Dmc, DIc);
  small_mm_k<<<cdiv(Bc * Lc,  TB), TB, 0, stream>>>(x_key, keys_w, keys0, Bc, Dmc, Lc);

  // ---- y = res, then accumulate the scans ----
  init_y_k<<<cdiv(Mc * DIc, TB), TB, 0, stream>>>(xz, y);

  dim3 tgrid(DIc / TB, Bc);
  temporal_scan_k<<<tgrid, TB, 0, stream>>>(delta,   p1, A_log,   Dv,  u, kp1, key2_w, y, 0);
  temporal_scan_k<<<tgrid, TB, 0, stream>>>(delta_b, p2, A_b_log, D_b, u, kp1, key2_w, y, 1);

  rowsum_k<<<Mc, TB, 0, stream>>>(delta_s, u, Ssum, Tsum);
  // spatial scan finalizes y and emits f16 directly
  spatial_scan_k<<<Mc / TB, TB, 0, stream>>>(Ssum, Tsum, p3, A_s_log, D_s, keys0, u, y, y16);

  // ---- out = y @ out_proj_w ----
  gemm(y16, wout16, out, Mc, Dmc, DIc, nullptr, 0);
}
